// CoreFoldingModule_60730837566357
// MI455X (gfx1250) — matrix operand order repaired and checked by reading, not compiled, not run on hardware
//
#include <hip/hip_runtime.h>
#include <hip/hip_bf16.h>

#define NRES 384
#define CPAIR 128
#define NP (NRES * NRES)   // 147456 pair rows; also N*C for single
#define NE 16384
#define NH 12

typedef __attribute__((ext_vector_type(16))) __bf16 v16bf;
typedef __attribute__((ext_vector_type(8)))  __bf16 v8bf;
typedef __attribute__((ext_vector_type(8)))  float  v8f;

__device__ __forceinline__ float actf(float x, int act) {
  if (act == 1) return x > 0.f ? x : 0.f;                  // relu
  if (act == 2) return 1.f / (1.f + __expf(-x));           // sigmoid
  if (act == 3) return x / (1.f + __expf(-x));             // silu
  return x;
}

#define BCOL 40   // bf16 elements per staged column (32 + pad, 80B => 16B-aligned)

// Stage one 32K x 64N bf16 B-tile into LDS via async-to-LDS b128 ops.
// B is pre-transposed (column-major: Bt[n*ldbt + k]), so both the global source
// and the LDS destination are contiguous per column: 256 x 16B transfers.
__device__ __forceinline__ void stage_async_bf16(const __bf16* __restrict__ Bt, int ldbt,
                                                 int n0, int N, int k0,
                                                 __bf16* dstbase, int tid)
{
#pragma unroll
  for (int it = 0; it < 2; ++it) {
    int s = tid + it * 128;        // 256 chunks of 8 bf16
    int col = s & 63;
    int kc  = s >> 6;              // 0..3
    int cc = n0 + col; if (cc >= N) cc = N - 1;
    const __bf16* gp = Bt + (long long)cc * ldbt + k0 + kc * 8;
    unsigned dst = (unsigned)(unsigned long long)(dstbase + col * BCOL + kc * 8);
    asm volatile("global_load_async_to_lds_b128 %0, %1, off"
                 :: "v"(dst), "v"(gp) : "memory");
  }
}

__device__ __forceinline__ v16bf load_afrag(const float* rowp, int k0, int hi)
{
  const float4* ap4 = (const float4*)(rowp + k0 + hi * 8);
  float4 a0 = ap4[0];   // K +0..3
  float4 a1 = ap4[1];   // K +4..7
  float4 a2 = ap4[4];   // K +16..19
  float4 a3 = ap4[5];   // K +20..23
  v16bf af;
  af[0]=(__bf16)a0.x; af[1]=(__bf16)a0.y; af[2]=(__bf16)a0.z; af[3]=(__bf16)a0.w;
  af[4]=(__bf16)a1.x; af[5]=(__bf16)a1.y; af[6]=(__bf16)a1.z; af[7]=(__bf16)a1.w;
  af[8]=(__bf16)a2.x; af[9]=(__bf16)a2.y; af[10]=(__bf16)a2.z; af[11]=(__bf16)a2.w;
  af[12]=(__bf16)a3.x; af[13]=(__bf16)a3.y; af[14]=(__bf16)a3.z; af[15]=(__bf16)a3.w;
  return af;
}

// B fragment: two 8-bf16 runs per lane (K 8h..8h+7, 16+8h..16+8h+7) = two ds_load_b128, no conversion.
__device__ __forceinline__ v16bf load_bfrag(const __bf16* bbase, int t, int lo, int hi)
{
  const __bf16* bp = bbase + (t * 16 + lo) * BCOL + hi * 8;
  v8bf h0 = *(const v8bf*)(bp);
  v8bf h1 = *(const v8bf*)(bp + 16);
  return __builtin_shufflevector(h0, h1, 0,1,2,3,4,5,6,7,8,9,10,11,12,13,14,15);
}

// C = act(A@B + bias [+ C_prev]) ; A f32 [M,K] lda ; B pre-transposed bf16
// (column-major, column stride ldbt = K) ; C f32 [M,N] ldc.
// blockDim=128 (4 waves). Block tile 128x64; each wave 32 rows x 64 cols
// (8 WMMAs / K-step). B double-buffered in LDS via async-to-LDS b128.
__global__ void gemm_bf16_wmma(const float* __restrict__ A, int lda, long long asz,
                               const __bf16* __restrict__ Bt, int ldbt, long long bsz,
                               const float* __restrict__ bias,
                               float* __restrict__ C, int ldc, long long csz,
                               int M, int N, int K, int act, int acc)
{
  __shared__ __bf16 ldsB[2][64 * BCOL];
  const int tid  = threadIdx.x;
  const int wave = tid >> 5;
  const int lane = tid & 31;
  const int lo = lane & 15;          // column-in-tile / row-in-A-tile
  const int hi = lane >> 4;          // K-half (A/B), M-half (C/D)
  const int mw = blockIdx.x * 128 + wave * 32;   // wave's 32-row strip
  const int n0 = blockIdx.y * 64;
  A  += (long long)blockIdx.z * asz;
  Bt += (long long)blockIdx.z * bsz;
  C  += (long long)blockIdx.z * csz;

  int ar0 = mw + lo;      if (ar0 >= M) ar0 = M - 1;
  int ar1 = mw + 16 + lo; if (ar1 >= M) ar1 = M - 1;
  const float* arow0 = A + (long long)ar0 * lda;
  const float* arow1 = A + (long long)ar1 * lda;

  v8f accv[2][4];
#pragma unroll
  for (int h01 = 0; h01 < 2; ++h01) {
#pragma unroll
    for (int t = 0; t < 4; ++t) {
      int col = n0 + t * 16 + lo;
#pragma unroll
      for (int r = 0; r < 8; ++r) {
        int row = mw + h01 * 16 + hi * 8 + r;
        accv[h01][t][r] = (acc && row < M && col < N) ? C[(long long)row * ldc + col] : 0.f;
      }
    }
  }

  const int KT = K >> 5;
  stage_async_bf16(Bt, ldbt, n0, N, 0, ldsB[0], tid);

  for (int kt = 0; kt < KT; ++kt) {
    asm volatile("s_wait_asynccnt 0" ::: "memory");   // tile kt arrived (own wave)
    __syncthreads();                                  // ... for all waves
    if (kt + 1 < KT)
      stage_async_bf16(Bt, ldbt, n0, N, (kt + 1) * 32, ldsB[(kt + 1) & 1], tid);

    int k0 = kt * 32;
    __builtin_prefetch(arow0 + k0 + 32, 0, 1);        // next A tile
    v16bf af0 = load_afrag(arow0, k0, hi);
    v16bf af1 = load_afrag(arow1, k0, hi);
    const __bf16* bbase = ldsB[kt & 1];
#pragma unroll
    for (int t = 0; t < 4; ++t) {
      v16bf bfr = load_bfrag(bbase, t, lo, hi);
      accv[0][t] = __builtin_amdgcn_wmma_f32_16x16x32_bf16(
          false, af0, false, bfr, (short)0, accv[0][t], false, false);
      accv[1][t] = __builtin_amdgcn_wmma_f32_16x16x32_bf16(
          false, af1, false, bfr, (short)0, accv[1][t], false, false);
    }
  }

#pragma unroll
  for (int h01 = 0; h01 < 2; ++h01) {
#pragma unroll
    for (int t = 0; t < 4; ++t) {
      int col = n0 + t * 16 + lo;
      if (col >= N) continue;
      float bv = bias ? bias[col] : 0.f;
#pragma unroll
      for (int r = 0; r < 8; ++r) {
        int row = mw + h01 * 16 + hi * 8 + r;
        if (row < M)
          C[(long long)row * ldc + col] = actf(accv[h01][t][r] + bv, act);
      }
    }
  }
}

// wt[n*K + k] = (bf16) w[k*N + n]   (transpose + convert, output-contiguous)
__global__ void convert_wT(const float* __restrict__ w, __bf16* __restrict__ wt,
                           int K, int N)
{
  long long idx = (long long)blockIdx.x * blockDim.x + threadIdx.x;
  if (idx >= (long long)K * N) return;
  int k = (int)(idx % K);
  int n = (int)(idx / K);
  wt[idx] = (__bf16)w[(long long)k * N + n];
}

// y[row,:] = LayerNorm(a[row,:] + resid[row,:]) * g + b     (one wave per row)
__global__ void add_ln(const float* __restrict__ a, const float* __restrict__ resid,
                       const float* __restrict__ g, const float* __restrict__ bet,
                       float* __restrict__ y, int rows, int D)
{
  int wave = threadIdx.x >> 5, lane = threadIdx.x & 31;
  int row = blockIdx.x * (blockDim.x >> 5) + wave;
  if (row >= rows) return;
  const float* pa = a + (long long)row * D;
  const float* pb = resid + (long long)row * D;
  float s = 0.f;
  for (int i = lane; i < D; i += 32) s += pa[i] + pb[i];
  for (int o = 16; o; o >>= 1) s += __shfl_xor(s, o, 32);
  float mean = s / D;
  float v = 0.f;
  for (int i = lane; i < D; i += 32) { float d = pa[i] + pb[i] - mean; v += d * d; }
  for (int o = 16; o; o >>= 1) v += __shfl_xor(v, o, 32);
  float inv = rsqrtf(v / D + 1e-5f);
  for (int i = lane; i < D; i += 32) {
    float d = (pa[i] + pb[i] - mean) * inv;
    y[(long long)row * D + i] = d * g[i] + bet[i];
  }
}

// rs[i,k] = sum_j R[i,j,k]
__global__ void rowsum_j(const float* __restrict__ R, float* __restrict__ rs)
{
  int idx = blockIdx.x * blockDim.x + threadIdx.x;
  if (idx >= NRES * CPAIR) return;
  int i = idx >> 7, k = idx & 127;
  const float* p = R + (long long)i * NRES * CPAIR + k;
  float s = 0.f;
  for (int j = 0; j < NRES; ++j) s += p[(long long)j * CPAIR];
  rs[idx] = s;
}

// T = L * rs(i,k) * G   (G already sigmoid'ed)
__global__ void tri_combine(const float* __restrict__ L, const float* __restrict__ G,
                            const float* __restrict__ rs, float* __restrict__ T)
{
  long long idx = (long long)blockIdx.x * blockDim.x + threadIdx.x;
  if (idx >= (long long)NP * CPAIR) return;
  int k = (int)(idx & 127);
  long long ij = idx >> 7;
  int i = (int)(ij / NRES);
  T[idx] = L[idx] * rs[i * CPAIR + k] * G[idx];
}

// q_g[n,h,c,p] = sum_a pts[n,h,a,p] * rot[n,a,c]; also sq[n,h]=sum(q_g^2)
__global__ void rot_points(const float* __restrict__ pts, const float* __restrict__ rot,
                           float* __restrict__ gout, float* __restrict__ ssum)
{
  int nh = blockIdx.x * blockDim.x + threadIdx.x;
  if (nh >= NRES * NH) return;
  int n = nh / NH, h = nh % NH;
  const float* pp = pts + (long long)n * (NH * 12) + h * 12;  // [a*4+p]
  const float* rr = rot + (long long)n * 9;                   // [a*3+c]
  float* go = gout + (long long)nh * 12;                      // [c*4+p]
  float s = 0.f;
  for (int c = 0; c < 3; ++c)
    for (int p = 0; p < 4; ++p) {
      float v = pp[0 * 4 + p] * rr[0 * 3 + c] + pp[1 * 4 + p] * rr[1 * 3 + c]
              + pp[2 * 4 + p] * rr[2 * 3 + c];
      go[c * 4 + p] = v;
      s += v * v;
    }
  ssum[nh] = s;
}

// attn[h,i,j] = qk*scale + pair_bias + point term
__global__ void ipa_logits(const float* __restrict__ q, const float* __restrict__ k,
                           const float* __restrict__ pb,
                           const float* __restrict__ qg, const float* __restrict__ kg,
                           const float* __restrict__ sq, const float* __restrict__ sk,
                           float* __restrict__ attn)
{
  long long idx = (long long)blockIdx.x * blockDim.x + threadIdx.x;
  if (idx >= (long long)NH * NP) return;
  int j = (int)(idx % NRES);
  int i = (int)((idx / NRES) % NRES);
  int h = (int)(idx / NP);
  const float scale = rsqrtf(32.f);
  const float* qi = q + (long long)i * 384 + h * 32;
  const float* kj = k + (long long)j * 384 + h * 32;
  float dq = 0.f;
  for (int c = 0; c < 32; ++c) dq += qi[c] * kj[c];
  const float* qgp = qg + ((long long)i * NH + h) * 12;
  const float* kgp = kg + ((long long)j * NH + h) * 12;
  float dp = 0.f;
  for (int c = 0; c < 12; ++c) dp += qgp[c] * kgp[c];
  float pt = -0.5f * (sq[i * NH + h] + sk[j * NH + h] - 2.f * dp) * scale;
  attn[idx] = dq * scale + pb[((long long)i * NRES + j) * NH + h] + pt;
}

__global__ void softmax_row(float* __restrict__ attn, int Ncols)
{
  __shared__ float red[256];
  long long base = (long long)blockIdx.x * Ncols;
  float m = -1e30f;
  for (int j = threadIdx.x; j < Ncols; j += 256) m = fmaxf(m, attn[base + j]);
  red[threadIdx.x] = m; __syncthreads();
  for (int s = 128; s; s >>= 1) {
    if ((int)threadIdx.x < s) red[threadIdx.x] = fmaxf(red[threadIdx.x], red[threadIdx.x + s]);
    __syncthreads();
  }
  m = red[0]; __syncthreads();
  float sum = 0.f;
  for (int j = threadIdx.x; j < Ncols; j += 256) {
    float e = __expf(attn[base + j] - m);
    attn[base + j] = e; sum += e;
  }
  red[threadIdx.x] = sum; __syncthreads();
  for (int s = 128; s; s >>= 1) {
    if ((int)threadIdx.x < s) red[threadIdx.x] += red[threadIdx.x + s];
    __syncthreads();
  }
  float inv = 1.f / red[0];
  for (int j = threadIdx.x; j < Ncols; j += 256) attn[base + j] *= inv;
}

// ea1[e,c] = silu(d[e]*we1[c] + be1[c])
__global__ void edge_embed1(const float* __restrict__ d, const float* __restrict__ w,
                            const float* __restrict__ b, float* __restrict__ o)
{
  long long idx = (long long)blockIdx.x * blockDim.x + threadIdx.x;
  if (idx >= (long long)NE * CPAIR) return;
  int e = (int)(idx >> 7), c = (int)(idx & 127);
  float x = d[e] * w[c] + b[c];
  o[idx] = x / (1.f + __expf(-x));
}

// mi[e,:] = concat(h[src], h[dst], ea[e])
__global__ void gather_mi(const float* __restrict__ h, const float* __restrict__ ea,
                          const int* __restrict__ eidx, float* __restrict__ mi)
{
  long long idx = (long long)blockIdx.x * blockDim.x + threadIdx.x;
  if (idx >= (long long)NE * 896) return;
  int c = (int)(idx % 896);
  int e = (int)(idx / 896);
  int src = eidx[e], dst = eidx[NE + e];
  float v;
  if (c < 384)      v = h[(long long)src * 384 + c];
  else if (c < 768) v = h[(long long)dst * 384 + (c - 384)];
  else              v = ea[(long long)e * CPAIR + (c - 768)];
  mi[idx] = v;
}

// cw[e] = dot(c1[e,:], wc2)
__global__ void coord_w(const float* __restrict__ c1, const float* __restrict__ w,
                        float* __restrict__ cw)
{
  int e = blockIdx.x * blockDim.x + threadIdx.x;
  if (e >= NE) return;
  const float* p = c1 + (long long)e * 256;
  float s = 0.f;
  for (int c = 0; c < 256; ++c) s += p[c] * w[c];
  cw[e] = s;
}

__global__ void scatter_h(const float* __restrict__ m, const int* __restrict__ eidx,
                          float* __restrict__ hout)
{
  long long idx = (long long)blockIdx.x * blockDim.x + threadIdx.x;
  if (idx >= (long long)NE * 384) return;
  int c = (int)(idx % 384);
  int e = (int)(idx / 384);
  int dst = eidx[NE + e];
  atomicAdd(&hout[(long long)dst * 384 + c], m[idx]);
}

__global__ void scatter_x(const float* __restrict__ cw, const float* __restrict__ coords,
                          const int* __restrict__ eidx, float* __restrict__ xout)
{
  int e = blockIdx.x * blockDim.x + threadIdx.x;
  if (e >= NE) return;
  int src = eidx[e], dst = eidx[NE + e];
  float w = cw[e];
  for (int d = 0; d < 3; ++d)
    atomicAdd(&xout[dst * 3 + d], w * (coords[src * 3 + d] - coords[dst * 3 + d]));
}

extern "C" void kernel_launch(void* const* d_in, const int* in_sizes, int n_in,
                              void* d_out, int out_size, void* d_ws, size_t ws_size,
                              hipStream_t stream)
{
  (void)in_sizes; (void)n_in; (void)out_size; (void)ws_size;
  const float* single = (const float*)d_in[0];
  const float* pairIn = (const float*)d_in[1];
  const float* coords = (const float*)d_in[2];
  const float* rot    = (const float*)d_in[3];
  const int*   eidx   = (const int*)d_in[5];
  const float* edist  = (const float*)d_in[6];
#define PW(i) ((const float*)d_in[(i)])
  // params flat order:
  // tri_out: 7 wl 8 bl 9 wr 10 br 11 wg 12 bg 13 wo 14 bo 15 we 16 be 17 g 18 bn
  // tri_in : 19 wl 20 bl 21 wr 22 br 23 wg 24 bg 25 wo 26 bo 27 g 28 bn
  // trans  : 29 w1 30 b1 31 w2 32 b2 33 g 34 bn
  // ipa    : 35 wq 36 bq 37 wk 38 bk 39 wv 40 bv 41 wp 42 bp 43 wpq 44 bpq 45 wpk 46 bpk 47 wo 48 bo 49 g 50 bn
  // egnn   : 51 we1 52 be1 53 we2 54 be2 55 wn1 56 bn1 57 wn2 58 bn2 59 wc1 60 bc1 61 wc2

  float* ws = (float*)d_ws;
  const long long NPC = (long long)NP * CPAIR;
  float* pairA = ws;
  float* bufL  = ws + NPC;
  float* bufR  = ws + 2 * NPC;
  float* bufG  = ws + 3 * NPC;
  float* bufT  = ws + 4 * NPC;
  float* rsbuf = ws + 5 * NPC;
  float* big   = rsbuf + NRES * CPAIR;           // NP*512 floats (transition hidden, reused by EGNN)
  float* pbuf  = big + (long long)NP * 512;      // NP*12
  float* attnb = pbuf + (long long)NP * NH;      // NH*NP
  float* qb = attnb + (long long)NH * NP;
  float* kb = qb + NRES * 384;
  float* vb = kb + NRES * 384;
  float* ob = vb + NRES * 384;
  float* sb = ob + NRES * 384;
  float* singleS = sb + NRES * 384;
  float* qpts = singleS + NRES * 384;
  float* kpts = qpts + NRES * 144;
  float* qg = kpts + NRES * 144;
  float* kg = qg + NRES * 144;
  float* sq = kg + NRES * 144;
  float* sk = sq + NRES * NH;
  // EGNN buffers aliased into `big` (transition done by then)
  float* mi  = big;
  float* m1  = mi + (long long)NE * 896;
  float* mm  = m1 + (long long)NE * 256;
  float* c1  = mm + (long long)NE * 384;
  float* ea1 = c1 + (long long)NE * 256;
  float* ea  = ea1 + (long long)NE * CPAIR;
  float* cw  = ea + (long long)NE * CPAIR;

  // bf16 transposed-weight arena (after all f32 buffers, 256B aligned)
  unsigned long long wb = (unsigned long long)(sk + NRES * NH);
  wb = (wb + 255ULL) & ~255ULL;
  __bf16* wbf = (__bf16*)wb;
  long long woff = 0;
  auto cvtT = [&](const float* w, int K, int N) -> const __bf16* {
    __bf16* dst = wbf + woff;
    long long n = (long long)K * N;
    convert_wT<<<(int)((n + 255) / 256), 256, 0, stream>>>(w, dst, K, N);
    woff += (n + 127) & ~127LL;   // keep 256B alignment
    return dst;
  };

  // pre-convert all GEMM B-operands (weights) to transposed bf16
  const __bf16 *wlT0 = cvtT(PW(7), 128, 128),  *weT0 = cvtT(PW(15), 128, 128);
  const __bf16 *wrT0 = cvtT(PW(9), 128, 128),  *wgT0 = cvtT(PW(11), 128, 128);
  const __bf16 *woT0 = cvtT(PW(13), 128, 128);
  const __bf16 *wlT1 = cvtT(PW(19), 128, 128), *wrT1 = cvtT(PW(21), 128, 128);
  const __bf16 *wgT1 = cvtT(PW(23), 128, 128), *woT1 = cvtT(PW(25), 128, 128);
  const __bf16 *w1T  = cvtT(PW(29), 128, 512), *w2T  = cvtT(PW(31), 512, 128);
  const __bf16 *wqT  = cvtT(PW(35), 384, 384), *wkT  = cvtT(PW(37), 384, 384);
  const __bf16 *wvT  = cvtT(PW(39), 384, 384), *wpT  = cvtT(PW(41), 128, 12);
  const __bf16 *wpqT = cvtT(PW(43), 384, 144), *wpkT = cvtT(PW(45), 384, 144);
  const __bf16 *woiT = cvtT(PW(47), 384, 384);
  const __bf16 *we2T = cvtT(PW(53), 128, 128), *wn1T = cvtT(PW(55), 896, 256);
  const __bf16 *wn2T = cvtT(PW(57), 256, 384), *wc1T = cvtT(PW(59), 896, 256);
  __bf16* vbT = wbf + woff;  woff += (long long)NRES * 384;   // attn@v B (activation)

  auto gemm = [&](const float* A, int lda, long long asz,
                  const __bf16* Bt, int ldbt, long long bsz, const float* bias,
                  float* C, int ldc, long long csz,
                  int M, int N, int K, int act, int acc, int batch) {
    dim3 grid((M + 127) / 128, (N + 63) / 64, batch);
    hipLaunchKernelGGL(gemm_bf16_wmma, grid, dim3(128), 0, stream,
                       A, lda, asz, Bt, ldbt, bsz, bias, C, ldc, csz, M, N, K, act, acc);
  };

  auto tri = [&](const float* pin,
                 const __bf16* wlT, const float* bl, const __bf16* weT, const float* be,
                 const __bf16* wrT, const float* br, const __bf16* wgT, const float* bg,
                 const __bf16* woT, const float* bo,
                 const float* gg, const float* bn, float* pout) {
    gemm(pin, 128, 0, wlT, 128, 0, bl, bufL, 128, 0, NP, 128, 128, 0, 0, 1);
    if (weT)
      gemm(pin, 128, 0, weT, 128, 0, be, bufL, 128, 0, NP, 128, 128, 0, 1, 1);
    gemm(pin, 128, 0, wrT, 128, 0, br, bufR, 128, 0, NP, 128, 128, 0, 0, 1);
    gemm(pin, 128, 0, wgT, 128, 0, bg, bufG, 128, 0, NP, 128, 128, 2, 0, 1);
    rowsum_j<<<(NRES * CPAIR + 255) / 256, 256, 0, stream>>>(bufR, rsbuf);
    tri_combine<<<(int)((NPC + 255) / 256), 256, 0, stream>>>(bufL, bufG, rsbuf, bufT);
    gemm(bufT, 128, 0, woT, 128, 0, bo, bufR, 128, 0, NP, 128, 128, 0, 0, 1);
    add_ln<<<(NP + 3) / 4, 128, 0, stream>>>(bufR, pin, gg, bn, pout, NP, 128);
  };

  // ---- triangle updates + transition on pair ----
  tri(pairIn, wlT0, PW(8), weT0, PW(16), wrT0, PW(10), wgT0, PW(12), woT0, PW(14),
      PW(17), PW(18), pairA);
  tri(pairA, wlT1, PW(20), nullptr, nullptr, wrT1, PW(22), wgT1, PW(24), woT1, PW(26),
      PW(27), PW(28), pairA);
  gemm(pairA, 128, 0, w1T, 128, 0, PW(30), big, 512, 0, NP, 512, 128, 1, 0, 1);
  gemm(big, 512, 0, w2T, 512, 0, PW(32), bufT, 128, 0, NP, 128, 512, 0, 0, 1);
  add_ln<<<(NP + 3) / 4, 128, 0, stream>>>(bufT, pairA, PW(33), PW(34), pairA, NP, 128);

  // ---- IPA ----
  gemm(single, 384, 0, wqT, 384, 0, PW(36), qb, 384, 0, NRES, 384, 384, 0, 0, 1);
  gemm(single, 384, 0, wkT, 384, 0, PW(38), kb, 384, 0, NRES, 384, 384, 0, 0, 1);
  gemm(single, 384, 0, wvT, 384, 0, PW(40), vb, 384, 0, NRES, 384, 384, 0, 0, 1);
  gemm(pairA, 128, 0, wpT, 128, 0, PW(42), pbuf, 12, 0, NP, 12, 128, 0, 0, 1);
  gemm(single, 384, 0, wpqT, 384, 0, PW(44), qpts, 144, 0, NRES, 144, 384, 0, 0, 1);
  gemm(single, 384, 0, wpkT, 384, 0, PW(46), kpts, 144, 0, NRES, 144, 384, 0, 0, 1);
  rot_points<<<(NRES * NH + 255) / 256, 256, 0, stream>>>(qpts, rot, qg, sq);
  rot_points<<<(NRES * NH + 255) / 256, 256, 0, stream>>>(kpts, rot, kg, sk);
  ipa_logits<<<(int)(((long long)NH * NP + 255) / 256), 256, 0, stream>>>(
      qb, kb, pbuf, qg, kg, sq, sk, attnb);
  softmax_row<<<NH * NRES, 256, 0, stream>>>(attnb, NRES);
  // out[i, h*32+c] = attn_h @ v_h : convert v to transposed bf16, batch over heads
  convert_wT<<<(NRES * 384 + 255) / 256, 256, 0, stream>>>(vb, vbT, 384, 384);
  gemm(attnb, 384, (long long)NP, vbT, 384, 32LL * 384, nullptr,
       ob, 384, 32, NRES, 32, 384, 0, 0, NH);
  gemm(ob, 384, 0, woiT, 384, 0, PW(48), sb, 384, 0, NRES, 384, 384, 0, 0, 1);
  add_ln<<<(NRES + 3) / 4, 128, 0, stream>>>(sb, single, PW(49), PW(50), singleS, NRES, 384);

  // ---- EGNN ----
  edge_embed1<<<(NE * CPAIR + 255) / 256, 256, 0, stream>>>(edist, PW(51), PW(52), ea1);
  gemm(ea1, 128, 0, we2T, 128, 0, PW(54), ea, 128, 0, NE, 128, 128, 0, 0, 1);
  gather_mi<<<(int)(((long long)NE * 896 + 255) / 256), 256, 0, stream>>>(singleS, ea, eidx, mi);
  gemm(mi, 896, 0, wn1T, 896, 0, PW(56), m1, 256, 0, NE, 256, 896, 3, 0, 1);
  gemm(m1, 256, 0, wn2T, 256, 0, PW(58), mm, 384, 0, NE, 384, 256, 0, 0, 1);
  gemm(mi, 896, 0, wc1T, 896, 0, PW(60), c1, 256, 0, NE, 256, 896, 3, 0, 1);
  coord_w<<<(NE + 255) / 256, 256, 0, stream>>>(c1, PW(61), cw);

  float* outH = (float*)d_out;
  float* outX = outH + NP;  // h is [1,384,384] = 147456 elems, then x [384,3]
  hipMemcpyAsync(outH, singleS, sizeof(float) * NRES * 384, hipMemcpyDeviceToDevice, stream);
  hipMemcpyAsync(outX, coords, sizeof(float) * NRES * 3, hipMemcpyDeviceToDevice, stream);
  scatter_h<<<(int)(((long long)NE * 384 + 255) / 256), 256, 0, stream>>>(mm, eidx, outH);
  scatter_x<<<(NE + 255) / 256, 256, 0, stream>>>(cw, coords, eidx, outX);
}
